// FoundationalTimeSeriesModel_63359357550973
// MI455X (gfx1250) — matrix operand order repaired and checked by reading, not compiled, not run on hardware
//
#include <hip/hip_runtime.h>
#include <math.h>

// ---------------- constants (match reference) ----------------
constexpr int kB  = 16;     // batch
constexpr int kS  = 512;    // sequence
constexpr int kM  = 32;     // sensors
constexpr int kC  = 64;     // tcn out dim (== proj dim P)
constexpr int kT  = kS + 1; // 513 (cls + seq)
constexpr int kN  = kB * kM;// 512 samples / tokens
constexpr int kD  = 128;    // d_model
constexpr int kH  = 8;      // heads
constexpr int kE  = 16;     // experts
constexpr int kEH = 256;    // expert hidden
constexpr int kMO = 128;    // expert out
constexpr int kKW = 192;    // conv GEMM K = Cin*3

typedef __attribute__((ext_vector_type(16))) _Float16 v16h;
typedef __attribute__((ext_vector_type(8)))  float    v8f;
typedef __attribute__((ext_vector_type(4)))  unsigned int u32x4;
typedef __attribute__((ext_vector_type(8)))  int i32x8;
typedef __attribute__((ext_vector_type(4)))  int i32x4;

#define WMMA16(a, b, c) \
  __builtin_amdgcn_wmma_f32_16x16x32_f16(false, (a), false, (b), (short)0, (c), false, false)

#if __has_builtin(__builtin_amdgcn_tensor_load_to_lds) && __has_builtin(__builtin_amdgcn_s_wait_tensorcnt)
#define USE_TDM 1
#else
#define USE_TDM 0
#endif

// ---------------- WMMA fragment helpers (wave32; contiguous k order, same for A and B) ----------------
// A 16x32 f16 tile: lanes 0-15 row=lane, k=k0+{0..15}; lanes 16-31 row=lane-16,
// k=k0+{16..31}. Unconditional contiguous loads.
__device__ __forceinline__ v16h frag_a16(const _Float16* __restrict__ A, int lda, int m0, int k0) {
  int lane = threadIdx.x & 31;
  const _Float16* p = A + (size_t)(m0 + (lane & 15)) * lda + k0 + ((lane & 16) ? 16 : 0);
  v16h a;
#pragma unroll
  for (int i = 0; i < 16; ++i) a[i] = p[i];
  return a;
}

// B 32x16 tile from torch-style weight W[Npad][Kpad] (B[k][n] = W[n][k]).
__device__ __forceinline__ v16h frag_b16(const _Float16* __restrict__ W, int ldw, int n0, int k0) {
  int lane = threadIdx.x & 31;
  const _Float16* p = W + (size_t)(n0 + (lane & 15)) * ldw + k0 + ((lane & 16) ? 16 : 0);
  v16h b;
#pragma unroll
  for (int j = 0; j < 16; ++j) b[j] = p[j];
  return b;
}

// B fragment gathered from LDS conv tile Bls[64][twid] (dilated taps; left-pad select).
__device__ __forceinline__ v16h frag_b_tile(const _Float16* __restrict__ Bls, int twid,
                                            int t0, int tstart, int dil, int k0) {
  int lane = threadIdx.x & 31;
  int tt   = lane & 15;
  int kb   = k0 + ((lane & 16) ? 16 : 0);
  v16h b;
#pragma unroll
  for (int j = 0; j < 16; ++j) {
    int kk = kb + j;
    int k = kk >> 6, ci = kk & 63;
    int t = t0 + tt - (2 - k) * dil;
    int tloc = t - tstart;
    int tlc = tloc < 0 ? 0 : tloc;         // clamp address, unconditional ds load
    _Float16 v = Bls[ci * twid + tlc];
    b[j] = (t >= 0) ? v : (_Float16)0.f;   // value select for causal left-pad
  }
  return b;
}

#if USE_TDM
// Issue a 2D TENSOR_LOAD_TO_LDS: tile [64 rows][twid cols] of f16 from a row-major
// [64][kT] tensor at gaddr (tile start), row stride kT elems, into LDS at lds_off.
// D# layout per CDNA5 ISA §8.3/8.4 (group0: count/lds/global/type, group1: dims/strides).
__device__ __forceinline__ void tdm_load_tile(unsigned lds_off, unsigned long long gaddr,
                                              int twid) {
  u32x4 g0;
  g0[0] = 1u;                                               // count=1, no gather
  g0[1] = lds_off;                                          // lds_addr (bytes)
  g0[2] = (unsigned)(gaddr & 0xFFFFFFFFull);                // global_addr[31:0]
  g0[3] = (unsigned)((gaddr >> 32) & 0x01FFFFFFull) | (2u << 30); // addr[56:32] | type=2
  i32x8 g1;
  g1[0] = (1 << 16);                                        // data_size=2B; wg_mask=0
  g1[1] = (int)(((unsigned)kT & 0xFFFFu) << 16);            // tensor_dim0[15:0]
  g1[2] = (int)((((unsigned)kT >> 16) & 0xFFFFu) | ((unsigned)kC << 16)); // dim0 hi | dim1 lo
  g1[3] = (int)((unsigned)twid << 16);                      // dim1 hi(0) | tile_dim0
  g1[4] = kC;                                               // tile_dim1=64, tile_dim2=0
  g1[5] = kT;                                               // tensor_dim0_stride[31:0]
  g1[6] = 0;                                                // stride0 hi | stride1 lo
  g1[7] = 0;
  i32x4 z4; z4[0] = 0; z4[1] = 0; z4[2] = 0; z4[3] = 0;     // groups 2/3 unused (<=2D)
#if defined(__clang_major__) && (__clang_major__ >= 23)
  i32x8 z8; for (int i = 0; i < 8; ++i) z8[i] = 0;
  __builtin_amdgcn_tensor_load_to_lds(g0, g1, z4, z4, z8, 0);
#else
  __builtin_amdgcn_tensor_load_to_lds(g0, g1, z4, z4, 0);
#endif
}
#endif

// ---------------- generic WMMA GEMM ----------------
// C = act(A16[Mr,Kd] @ W16[Nc,Kd]^T + bias) (+ residual). Kd %32==0 (operands zero-
// padded), Mr %16==0, weight rows padded to x16. act: 0 none, 1 relu, 2 sigmoid.
__global__ void k_gemm(const _Float16* __restrict__ A, int lda,
                       const _Float16* __restrict__ W, int ldw,
                       const float* __restrict__ bias,
                       const float* __restrict__ residual,
                       float* __restrict__ Cf, _Float16* __restrict__ Ch, int ldc,
                       int Mr, int Nc, int Kd, int act) {
  int m0 = blockIdx.y * 16;
  int n0 = (blockIdx.x * 4 + (threadIdx.x >> 5)) * 16;
  if (n0 >= Nc) return;                       // uniform per wave: EXEC stays all-ones
  int lane  = threadIdx.x & 31;
  int col   = n0 + (lane & 15);
  int rbase = (lane & 16) ? 8 : 0;
  float bv = bias ? bias[col < Nc ? col : Nc - 1] : 0.f;  // clamped, unconditional
  v8f c;
#pragma unroll
  for (int i = 0; i < 8; ++i) c[i] = bv;
  int ks = Kd >> 5;
  for (int s = 0; s < ks; ++s) {
    v16h a = frag_a16(A, lda, m0, s * 32);
    v16h b = frag_b16(W, ldw, n0, s * 32);
    c = WMMA16(a, b, c);
  }
#pragma unroll
  for (int i = 0; i < 8; ++i) {
    int row = m0 + rbase + i;
    if (col < Nc) {
      float v = c[i];
      if (act == 1) v = fmaxf(v, 0.f);
      else if (act == 2) v = 1.f / (1.f + expf(-v));
      if (residual) v += residual[(size_t)row * ldc + col];
      if (Cf) Cf[(size_t)row * ldc + col] = v;
      if (Ch) Ch[(size_t)row * ldc + col] = (_Float16)v;
    }
  }
}

// cast + zero-pad weight [Nrows,Kc] f32 -> [Npad,Kpad] f16
__global__ void k_wcast(const float* __restrict__ w, _Float16* __restrict__ o,
                        int Nrows, int Kc, int Npad, int Kpad) {
  int idx = blockIdx.x * 256 + threadIdx.x;
  if (idx >= Npad * Kpad) return;
  int n = idx / Kpad, k = idx % Kpad;
  float v = (n < Nrows && k < Kc) ? w[(size_t)n * Kc + k] : 0.f;
  o[idx] = (_Float16)v;
}

// ---------------- TCN causal dilated conv as implicit WMMA GEMM ----------------
// in/out f16 [N][64][T], Wf f16 [64][192] kk=k*64+ci. out = relu(conv + bias).
// Input tile [64][tstart..t0+15] staged to LDS via TDM (tensor_load_to_lds) when
// available, else cooperative vector staging; dilated taps gathered at frag build.
__global__ void k_tcn_conv(const _Float16* __restrict__ in, const _Float16* __restrict__ Wf,
                           const float* __restrict__ bias, _Float16* __restrict__ out, int dil) {
  __shared__ _Float16 Bls[kC * 48];   // max twid = 16 + 2*8 = 32 (<48)
  int n  = blockIdx.x;
  int t0 = blockIdx.y * 16;
  int tid = threadIdx.x; // 128 threads = 4 waves
  const _Float16* inb = in + (size_t)n * kC * kT;
  int tstart = t0 - 2 * dil; if (tstart < 0) tstart = 0;
  int twid = t0 + 16 - tstart;
#if USE_TDM
  if (tid < 32) {                      // one wave issues the DMA (EXEC-ignored op)
    unsigned lds_off = (unsigned)(unsigned long long)(const void*)Bls; // flat LDS -> as3 offset
    unsigned long long ga = (unsigned long long)(size_t)(const void*)inb +
                            (unsigned long long)tstart * 2ull;
    tdm_load_tile(lds_off, ga, twid);
    __builtin_amdgcn_s_wait_tensorcnt(0);
  }
  __syncthreads();
#else
  for (int idx = tid; idx < kC * twid; idx += 128) {
    int ci = idx / twid, tl = idx % twid;
    int t = tstart + tl;
    int tc = t >= kT ? kT - 1 : t;              // clamp address, unconditional load
    _Float16 v = inb[ci * kT + tc];
    Bls[ci * twid + tl] = (t < kT) ? v : (_Float16)0.f;
  }
  __syncthreads();
#endif
  int co0   = (tid >> 5) * 16;
  int lane  = tid & 31;
  int col   = lane & 15;
  int rbase = (lane & 16) ? 8 : 0;
  v8f c;
#pragma unroll
  for (int i = 0; i < 8; ++i) c[i] = bias[co0 + rbase + i];
#pragma unroll
  for (int s = 0; s < 6; ++s) {  // K = 192 = 6 * 32
    v16h a = frag_a16(Wf, kKW, co0, s * 32);
    v16h b = frag_b_tile(Bls, twid, t0, tstart, dil, s * 32);
    c = WMMA16(a, b, c);
  }
  int t = t0 + col;
#pragma unroll
  for (int i = 0; i < 8; ++i) {
    float v = fmaxf(c[i], 0.f);
    if (t < kT) out[(size_t)n * kC * kT + (co0 + rbase + i) * kT + t] = (_Float16)v;
  }
}

// conv weight [64,64,3] f32 -> [64][k*64+ci] f16
__global__ void k_conv_wcast(const float* __restrict__ w, _Float16* __restrict__ o) {
  int idx = blockIdx.x * 256 + threadIdx.x;
  if (idx >= 64 * kKW) return;
  int co = idx / kKW, kk = idx % kKW;
  int k = kk >> 6, ci = kk & 63;
  o[idx] = (_Float16)w[(co * 64 + ci) * 3 + k];
}

// ---------------- prep: input proj + cls + positional encoding -> f16 [N][64][T] ----------------
__global__ void k_prep(const float* __restrict__ x, const float* __restrict__ in_w,
                       const float* __restrict__ in_b, const float* __restrict__ cls,
                       _Float16* __restrict__ h) {
  long idx = (long)blockIdx.x * blockDim.x + threadIdx.x;
  const long total = (long)kN * kC * kT;
  if (idx >= total) return;
  int t = (int)(idx % kT);
  long rem = idx / kT;
  int p = (int)(rem % kC);
  long n = rem / kC;
  float v;
  if (t == 0) v = cls[p];
  else v = x[n * kS + (t - 1)] * in_w[p] + in_b[p];  // torch-faithful flat reshape
  int i2 = p & ~1;
  float div = expf(-(float)i2 * (logf(10000.f) / (float)kC));
  float ang = (float)t * div;
  v += (p & 1) ? cosf(ang) : sinf(ang);
  h[idx] = (_Float16)v;
}

__global__ void k_resrelu(const _Float16* __restrict__ o, _Float16* __restrict__ h, long total) {
  long idx = (long)blockIdx.x * blockDim.x + threadIdx.x;
  if (idx >= total) return;
  float v = (float)o[idx] + (float)h[idx];
  h[idx] = (_Float16)(v > 0.f ? v : 0.f);
}

// per-(n,t) LN stats over 64 channels
__global__ void k_lnstats(const _Float16* __restrict__ h, float* __restrict__ mu,
                          float* __restrict__ rsd) {
  int idx = blockIdx.x * blockDim.x + threadIdx.x;
  if (idx >= kN * kT) return;
  int t = idx % kT, n = idx / kT;
  const _Float16* p = h + (size_t)n * kC * kT + t;
  float s = 0.f, sq = 0.f;
  for (int c = 0; c < kC; ++c) { float v = (float)p[c * kT]; s += v; sq += v * v; }
  float m = s / (float)kC;
  mu[idx]  = m;
  rsd[idx] = rsqrtf(sq / (float)kC - m * m + 1e-5f);
}

// pooled[n][c] = mean_t LN(h)[n,t,c] * mask[n]  (f32 + f16 mirror)
__global__ void k_pooled(const _Float16* __restrict__ h, const float* __restrict__ mu,
                         const float* __restrict__ rsd, const float* __restrict__ g,
                         const float* __restrict__ b, const float* __restrict__ mask,
                         float* __restrict__ pooled, _Float16* __restrict__ pooled16) {
  int idx = blockIdx.x * blockDim.x + threadIdx.x;
  if (idx >= kN * kC) return;
  int c = idx % kC, n = idx / kC;
  const _Float16* p = h + (size_t)n * kC * kT + (size_t)c * kT;
  const float* mun = mu + n * kT;
  const float* rn  = rsd + n * kT;
  float s = 0.f;
  for (int t = 1; t < kT; ++t) s += ((float)p[t] - mun[t]) * rn[t];
  float v = ((s / (float)kS) * g[c] + b[c]) * mask[n];
  pooled[idx]   = v;
  pooled16[idx] = (_Float16)v;
}

__global__ void k_hcls(const _Float16* __restrict__ h, const float* __restrict__ mu,
                       const float* __restrict__ rsd, const float* __restrict__ g,
                       const float* __restrict__ b, float* __restrict__ hcls) {
  int idx = blockIdx.x * blockDim.x + threadIdx.x;
  if (idx >= kN * kC) return;
  int c = idx % kC, n = idx / kC;
  float v = (float)h[(size_t)n * kC * kT + (size_t)c * kT] - mu[n * kT];
  hcls[idx] = v * rsd[n * kT] * g[c] + b[c];
}

__global__ void k_addpos(const float* __restrict__ proj, const float* __restrict__ pos,
                         float* __restrict__ out) {
  int idx = blockIdx.x * blockDim.x + threadIdx.x;
  if (idx >= kN * kD) return;
  int d = idx % kD, n = idx / kD;
  out[idx] = proj[idx] + pos[(n % kM) * kD + d];
}

// row LayerNorm; optional per-row scale (mask); f32 and/or f16 outputs
__global__ void k_lnrows(const float* __restrict__ in, float* __restrict__ outf,
                         _Float16* __restrict__ outh, const float* __restrict__ g,
                         const float* __restrict__ b, const float* __restrict__ rowscale,
                         int R, int Dd) {
  int row = blockIdx.x * blockDim.x + threadIdx.x;
  if (row >= R) return;
  const float* x = in + (size_t)row * Dd;
  float mu = 0.f;
  for (int d = 0; d < Dd; ++d) mu += x[d];
  mu /= (float)Dd;
  float var = 0.f;
  for (int d = 0; d < Dd; ++d) { float t = x[d] - mu; var += t * t; }
  float rs = rsqrtf(var / (float)Dd + 1e-5f);
  float sc = rowscale ? rowscale[row] : 1.f;
  for (int d = 0; d < Dd; ++d) {
    float v = ((x[d] - mu) * rs * g[d] + b[d]) * sc;
    if (outf) outf[(size_t)row * Dd + d] = v;
    if (outh) outh[(size_t)row * Dd + d] = (_Float16)v;
  }
}

// attention: one block per (b,h), one thread per query row; writes f16 (GEMM input)
__global__ void k_attn(const float* __restrict__ qkv, const float* __restrict__ mask,
                       _Float16* __restrict__ o16) {
  int bh = blockIdx.x;
  int b = bh >> 3, h = bh & 7;
  int m = threadIdx.x; // 32
  const float* base = qkv + (size_t)(b * kM) * 384;
  float q[16];
#pragma unroll
  for (int d = 0; d < 16; ++d) q[d] = base[m * 384 + h * 16 + d];
  float sc[32];
  float mx = -1e30f;
  for (int k = 0; k < kM; ++k) {
    float s = 0.f;
#pragma unroll
    for (int d = 0; d < 16; ++d) s += q[d] * base[k * 384 + 128 + h * 16 + d];
    s *= 0.25f;
    if (mask[b * kM + k] == 0.f) s = -1e9f;
    sc[k] = s;
    mx = fmaxf(mx, s);
  }
  float ssum = 0.f;
  for (int k = 0; k < kM; ++k) { sc[k] = expf(sc[k] - mx); ssum += sc[k]; }
  float inv = 1.f / ssum;
#pragma unroll
  for (int d = 0; d < 16; ++d) {
    float acc = 0.f;
    for (int k = 0; k < kM; ++k) acc += sc[k] * base[k * 384 + 256 + h * 16 + d];
    o16[(size_t)(b * kM + m) * kD + h * 16 + d] = (_Float16)(acc * inv);
  }
}

// gate_g16[b][0:128]=mean_ctx [128:256]=mean_proj [256:320]=mean_hcls ; meanctx16 mirror
__global__ void k_gate(const float* __restrict__ ctxm, const float* __restrict__ proj,
                       const float* __restrict__ hcls, const float* __restrict__ mask,
                       _Float16* __restrict__ gate_g16, _Float16* __restrict__ meanctx16) {
  int idx = blockIdx.x * blockDim.x + threadIdx.x;
  if (idx >= kB * 320) return;
  int b = idx / 320, j = idx % 320;
  float act = 0.f;
  for (int m = 0; m < kM; ++m) act += mask[b * kM + m];
  act = fmaxf(act, 1.f);
  float s = 0.f;
  if (j < 128) { for (int m = 0; m < kM; ++m) s += ctxm[(size_t)(b * kM + m) * kD + j]; }
  else if (j < 256) { int d = j - 128; for (int m = 0; m < kM; ++m) s += proj[(size_t)(b * kM + m) * kD + d] * mask[b * kM + m]; }
  else { int c = j - 256; for (int m = 0; m < kM; ++m) s += hcls[(size_t)(b * kM + m) * kC + c] * mask[b * kM + m]; }
  s /= act;
  gate_g16[idx] = (_Float16)s;
  if (j < 128) meanctx16[b * kD + j] = (_Float16)s;
}

// top-2 weighted expert combine; f32 and/or f16 outputs
__global__ void k_moe_combine(const float* __restrict__ logits, const float* __restrict__ allo,
                              float* __restrict__ yf, _Float16* __restrict__ yh) {
  int row = blockIdx.x, col = threadIdx.x; // 128 cols
  const float* lg = logits + row * kE;
  int i0 = 0; float v0 = lg[0];
  for (int e = 1; e < kE; ++e) if (lg[e] > v0) { v0 = lg[e]; i0 = e; }
  int i1 = 0; float v1 = -1e30f;
  for (int e = 0; e < kE; ++e) if (e != i0 && lg[e] > v1) { v1 = lg[e]; i1 = e; }
  float e1 = expf(v1 - v0);
  float inv = 1.f / (1.f + e1);
  float v = inv * (allo[(size_t)row * kE * kMO + i0 * kMO + col] +
                   e1 * allo[(size_t)row * kE * kMO + i1 * kMO + col]);
  if (yf) yf[(size_t)row * kMO + col] = v;
  if (yh) yh[(size_t)row * kMO + col] = (_Float16)v;
}

// load-balance + entropy for one gate (deterministic LDS reduction, one block)
__global__ void k_gate_stats(const float* __restrict__ logits, int Nr, float* __restrict__ out2) {
  __shared__ float sp[256 * kE];
  __shared__ float sc[256 * kE];
  __shared__ float se[256];
  int tid = threadIdx.x;
  float pacc[kE], cacc[kE];
  for (int e = 0; e < kE; ++e) { pacc[e] = 0.f; cacc[e] = 0.f; }
  float eacc = 0.f;
  for (int r = tid; r < Nr; r += 256) {
    const float* lg = logits + r * kE;
    float mx = lg[0];
    for (int e = 1; e < kE; ++e) mx = fmaxf(mx, lg[e]);
    float s = 0.f, ex[kE];
    for (int e = 0; e < kE; ++e) { ex[e] = expf(lg[e] - mx); s += ex[e]; }
    float inv = 1.f / s, lse = logf(s) + mx;
    float ent = 0.f;
    for (int e = 0; e < kE; ++e) { float p = ex[e] * inv; pacc[e] += p; ent -= p * (lg[e] - lse); }
    eacc += ent;
    int i0 = 0; float v0 = lg[0];
    for (int e = 1; e < kE; ++e) if (lg[e] > v0) { v0 = lg[e]; i0 = e; }
    int i1 = 0; float v1 = -1e30f;
    for (int e = 0; e < kE; ++e) if (e != i0 && lg[e] > v1) { v1 = lg[e]; i1 = e; }
    cacc[i0] += 1.f; cacc[i1] += 1.f;
  }
  for (int e = 0; e < kE; ++e) { sp[tid * kE + e] = pacc[e]; sc[tid * kE + e] = cacc[e]; }
  se[tid] = eacc;
  __syncthreads();
  if (tid == 0) {
    float ap[kE], cc[kE], esum = 0.f;
    for (int e = 0; e < kE; ++e) { ap[e] = 0.f; cc[e] = 0.f; }
    for (int t = 0; t < 256; ++t) {
      for (int e = 0; e < kE; ++e) { ap[e] += sp[t * kE + e]; cc[e] += sc[t * kE + e]; }
      esum += se[t];
    }
    float lb = 0.f;
    for (int e = 0; e < kE; ++e) lb += (ap[e] / (float)Nr) * (cc[e] / (float)(Nr * 2));
    out2[0] = (float)kE * lb;
    out2[1] = esum / (float)Nr;
  }
}

// pred head input concat -> f16 [512][320]
__global__ void k_predin(const float* __restrict__ pooled, const float* __restrict__ ctxm,
                         const float* __restrict__ y_f, _Float16* __restrict__ out16) {
  int idx = blockIdx.x * blockDim.x + threadIdx.x;
  if (idx >= kN * 320) return;
  int j = idx % 320, n = idx / 320, b = n >> 5;
  float v;
  if (j < 64)       v = pooled[n * kC + j];
  else if (j < 192) v = ctxm[(size_t)n * kD + (j - 64)];
  else              v = y_f[b * kMO + (j - 192)];
  out16[idx] = (_Float16)v;
}

// rca head input concat -> f16 [512][352] (K=321 zero-padded to 352)
__global__ void k_rcain(const float* __restrict__ pooled, const float* __restrict__ ctxm,
                        const float* __restrict__ y_r, const float* __restrict__ x,
                        const float* __restrict__ lastv, _Float16* __restrict__ out16) {
  int idx = blockIdx.x * blockDim.x + threadIdx.x;
  if (idx >= kN * 352) return;
  int j = idx % 352, n = idx / 352;
  float v;
  if (j < 64)       v = pooled[n * kC + j];
  else if (j < 192) v = ctxm[(size_t)n * kD + (j - 64)];
  else if (j < 320) v = y_r[(size_t)n * kMO + (j - 192)];
  else if (j == 320) {
    int b = n >> 5, m = n & 31;
    v = fabsf(x[((size_t)b * kS + (kS - 1)) * kM + m] - lastv[b * kM + m]);
  } else v = 0.f;
  out16[idx] = (_Float16)v;
}

__global__ void k_finalize(const float* __restrict__ a6, float* __restrict__ out) {
  out[0] = 0.01f * (a6[0] + a6[2] + a6[4]) + 0.001f * (a6[1] + a6[3] + a6[5]);
}

// ---------------- host launch ----------------
static inline void gemm(hipStream_t st, const _Float16* A, int lda, const _Float16* W, int ldw,
                        const float* bias, const float* res, float* Cf, _Float16* Ch, int ldc,
                        int M, int N, int K, int act) {
  dim3 g((N + 63) / 64, (M + 15) / 16);
  k_gemm<<<g, 128, 0, st>>>(A, lda, W, ldw, bias, res, Cf, Ch, ldc, M, N, K, act);
}

extern "C" void kernel_launch(void* const* d_in, const int* in_sizes, int n_in,
                              void* d_out, int out_size, void* d_ws, size_t ws_size,
                              hipStream_t stream) {
  (void)in_sizes; (void)n_in; (void)out_size; (void)ws_size;
  const float* x     = (const float*)d_in[0];
  const float* mask  = (const float*)d_in[1];
  const float* lastv = (const float*)d_in[2];
  auto PP = [&](int i) { return (const float*)d_in[3 + i]; };
  // param leaves in dict-insertion order
  const int IN_W=0, IN_B=1, CLS=2;                               // tcn leaves 3..18
  const int FN_G=19, FN_B=20, PROJ_W=21, PROJ_B=22, POS=23;      // layers 24..47 (12 each)
  const int ON_G=48, ON_B=49, GF_W=50, GF_B=51, GL_W=52, GL_B=53, GR_W=54, GR_B=55;
  const int EX_W1=56, EX_B1=57, EX_W2=58, EX_B2=59, PH_W=60, PH_B=61, HZ_W=62, HZ_B=63;
  const int RC_W=64, RC_B=65;

  // ---- workspace carve ----
  char* base = (char*)d_ws;
  size_t off = 0;
  auto alloc = [&](size_t bytes) -> char* {
    char* p = base + off; off += (bytes + 255) & ~(size_t)255; return p;
  };
  const long NT = (long)kN * kC * kT;
  _Float16* hA   = (_Float16*)alloc(NT * 2);
  _Float16* hB   = (_Float16*)alloc(NT * 2);
  _Float16* hC   = (_Float16*)alloc(NT * 2);
  _Float16* Wf   = (_Float16*)alloc((size_t)8 * 64 * kKW * 2);
  float* mu      = (float*)alloc((size_t)kN * kT * 4);
  float* rsd     = (float*)alloc((size_t)kN * kT * 4);
  float* pooled  = (float*)alloc((size_t)kN * kC * 4);
  _Float16* pooled16 = (_Float16*)alloc((size_t)kN * kC * 2);
  float* hcls    = (float*)alloc((size_t)kN * kC * 4);
  float* proj    = (float*)alloc((size_t)kN * kD * 4);
  float* tb0     = (float*)alloc((size_t)kN * kD * 4);
  float* tb1     = (float*)alloc((size_t)kN * kD * 4);
  _Float16* hbuf16   = (_Float16*)alloc((size_t)kN * kD * 2);
  float* qkv     = (float*)alloc((size_t)kN * 384 * 4);
  _Float16* attn16   = (_Float16*)alloc((size_t)kN * kD * 2);
  _Float16* ffn116   = (_Float16*)alloc((size_t)kN * 256 * 2);
  float* ctxm    = (float*)alloc((size_t)kN * kD * 4);
  _Float16* ctxm16   = (_Float16*)alloc((size_t)kN * kD * 2);
  _Float16* gate16   = (_Float16*)alloc((size_t)kB * 320 * 2);
  _Float16* mctx16   = (_Float16*)alloc((size_t)kB * kD * 2);
  float* lg_f    = (float*)alloc((size_t)kB * kE * 4);
  float* lg_l    = (float*)alloc((size_t)kB * kE * 4);
  float* lg_r    = (float*)alloc((size_t)kN * kE * 4);
  _Float16* hidf16   = (_Float16*)alloc((size_t)kB * kE * kEH * 2);
  _Float16* hidr16   = (_Float16*)alloc((size_t)kN * kE * kEH * 2);
  float* allo_f  = (float*)alloc((size_t)kB * kE * kMO * 4);
  float* allo_r  = (float*)alloc((size_t)kN * kE * kMO * 4);
  float* y_f     = (float*)alloc((size_t)kB * kMO * 4);
  _Float16* yfl16    = (_Float16*)alloc((size_t)kB * kMO * 2);
  float* y_r     = (float*)alloc((size_t)kN * kMO * 4);
  _Float16* predin16 = (_Float16*)alloc((size_t)kN * 320 * 2);
  _Float16* rcain16  = (_Float16*)alloc((size_t)kN * 352 * 2);
  float* aux6    = (float*)alloc(8 * 4);
  // f16 padded weights
  _Float16* projw16 = (_Float16*)alloc((size_t)128 * 64 * 2);
  _Float16* wqkv16[2], *wo16[2], *w116[2], *w216[2];
  for (int l = 0; l < 2; ++l) {
    wqkv16[l] = (_Float16*)alloc((size_t)384 * 128 * 2);
    wo16[l]   = (_Float16*)alloc((size_t)128 * 128 * 2);
    w116[l]   = (_Float16*)alloc((size_t)256 * 128 * 2);
    w216[l]   = (_Float16*)alloc((size_t)128 * 256 * 2);
  }
  _Float16* gfw16 = (_Float16*)alloc((size_t)16 * 320 * 2);
  _Float16* glw16 = (_Float16*)alloc((size_t)16 * 320 * 2);
  _Float16* grw16 = (_Float16*)alloc((size_t)16 * 128 * 2);
  _Float16* exw116 = (_Float16*)alloc((size_t)kE * kEH * kD * 2);
  _Float16* exw216 = (_Float16*)alloc((size_t)kE * kMO * kEH * 2);
  _Float16* phw16 = (_Float16*)alloc((size_t)16 * 320 * 2);
  _Float16* hzw16 = (_Float16*)alloc((size_t)64 * 128 * 2);
  _Float16* rcw16 = (_Float16*)alloc((size_t)16 * 352 * 2);

  float* out_pred = (float*)d_out;          // [512,8]
  float* out_haz  = (float*)d_out + 4096;   // [16,64]
  float* out_rca  = (float*)d_out + 5120;   // [512]
  float* out_aux  = (float*)d_out + 5632;   // [1]

  auto wcast = [&](const float* w, _Float16* o, int N, int K, int Np, int Kp) {
    k_wcast<<<(Np * Kp + 255) / 256, 256, 0, stream>>>(w, o, N, K, Np, Kp);
  };

  // ---- 0. weight casts (f16, padded) ----
  for (int i = 0; i < 4; ++i) {
    k_conv_wcast<<<(64 * kKW + 255) / 256, 256, 0, stream>>>(PP(3 + 4 * i + 0), Wf + (size_t)(2 * i) * 64 * kKW);
    k_conv_wcast<<<(64 * kKW + 255) / 256, 256, 0, stream>>>(PP(3 + 4 * i + 2), Wf + (size_t)(2 * i + 1) * 64 * kKW);
  }
  wcast(PP(PROJ_W), projw16, 128, 64, 128, 64);
  for (int l = 0; l < 2; ++l) {
    int L0 = 24 + 12 * l;
    wcast(PP(L0 + 0), wqkv16[l], 384, 128, 384, 128);
    wcast(PP(L0 + 2), wo16[l],   128, 128, 128, 128);
    wcast(PP(L0 + 8), w116[l],   256, 128, 256, 128);
    wcast(PP(L0 + 10), w216[l],  128, 256, 128, 256);
  }
  wcast(PP(GF_W), gfw16, 16, 320, 16, 320);
  wcast(PP(GL_W), glw16, 16, 320, 16, 320);
  wcast(PP(GR_W), grw16, 16, 128, 16, 128);
  wcast(PP(EX_W1), exw116, kE * kEH, kD, kE * kEH, kD);
  wcast(PP(EX_W2), exw216, kE * kMO, kEH, kE * kMO, kEH);
  wcast(PP(PH_W), phw16, 8, 320, 16, 320);
  wcast(PP(HZ_W), hzw16, 64, 128, 64, 128);
  wcast(PP(RC_W), rcw16, 1, 321, 16, 352);

  // ---- 1. input proj + cls + PE ----
  k_prep<<<(unsigned)((NT + 255) / 256), 256, 0, stream>>>(x, PP(IN_W), PP(IN_B), PP(CLS), hA);

  // ---- 2. TCN: 4 levels x (conv,conv,res) with WMMA + TDM staging ----
  dim3 cgrid(kN, (kT + 15) / 16);
  for (int i = 0; i < 4; ++i) {
    int dil = 1 << i;
    k_tcn_conv<<<cgrid, 128, 0, stream>>>(hA, Wf + (size_t)(2 * i) * 64 * kKW, PP(3 + 4 * i + 1), hB, dil);
    k_tcn_conv<<<cgrid, 128, 0, stream>>>(hB, Wf + (size_t)(2 * i + 1) * 64 * kKW, PP(3 + 4 * i + 3), hC, dil);
    k_resrelu<<<(unsigned)((NT + 255) / 256), 256, 0, stream>>>(hC, hA, NT);
  }

  // ---- 3. final-norm stats, pooled, h_cls ----
  k_lnstats<<<(kN * kT + 255) / 256, 256, 0, stream>>>(hA, mu, rsd);
  k_pooled<<<(kN * kC + 255) / 256, 256, 0, stream>>>(hA, mu, rsd, PP(FN_G), PP(FN_B), mask, pooled, pooled16);
  k_hcls<<<(kN * kC + 255) / 256, 256, 0, stream>>>(hA, mu, rsd, PP(FN_G), PP(FN_B), hcls);

  // ---- 4. projection + positional ----
  gemm(stream, pooled16, kC, projw16, kC, PP(PROJ_B), nullptr, proj, nullptr, kD, kN, kD, kC, 0);
  k_addpos<<<(kN * kD + 255) / 256, 256, 0, stream>>>(proj, PP(POS), tb0);

  // ---- 5. transformer encoder (2 layers, norm_first) ----
  for (int l = 0; l < 2; ++l) {
    int L0 = 24 + 12 * l;
    k_lnrows<<<(kN + 255) / 256, 256, 0, stream>>>(tb0, nullptr, hbuf16, PP(L0 + 4), PP(L0 + 5), nullptr, kN, kD);
    gemm(stream, hbuf16, kD, wqkv16[l], kD, PP(L0 + 1), nullptr, qkv, nullptr, 384, kN, 384, kD, 0);
    k_attn<<<kB * kH, kM, 0, stream>>>(qkv, mask, attn16);
    gemm(stream, attn16, kD, wo16[l], kD, PP(L0 + 3), tb0, tb1, nullptr, kD, kN, kD, kD, 0);
    k_lnrows<<<(kN + 255) / 256, 256, 0, stream>>>(tb1, nullptr, hbuf16, PP(L0 + 6), PP(L0 + 7), nullptr, kN, kD);
    gemm(stream, hbuf16, kD, w116[l], kD, PP(L0 + 9), nullptr, nullptr, ffn116, 256, kN, 256, kD, 1);
    gemm(stream, ffn116, 256, w216[l], 256, PP(L0 + 11), tb1, tb0, nullptr, kD, kN, kD, 256, 0);
  }

  // ---- 6. output norm (masked) + gate features ----
  k_lnrows<<<(kN + 255) / 256, 256, 0, stream>>>(tb0, ctxm, ctxm16, PP(ON_G), PP(ON_B), mask, kN, kD);
  k_gate<<<(kB * 320 + 255) / 256, 256, 0, stream>>>(ctxm, proj, hcls, mask, gate16, mctx16);

  // ---- 7. gate logits ----
  gemm(stream, gate16, 320, gfw16, 320, PP(GF_B), nullptr, lg_f, nullptr, kE, kB, kE, 320, 0);
  gemm(stream, gate16, 320, glw16, 320, PP(GL_B), nullptr, lg_l, nullptr, kE, kB, kE, 320, 0);
  gemm(stream, ctxm16, kD, grw16, kD, PP(GR_B), nullptr, lg_r, nullptr, kE, kN, kE, kD, 0);

  // ---- 8. experts (all E as in reference; hid/allo shared by f & fl gates) ----
  for (int e = 0; e < kE; ++e) {
    const _Float16* w1 = exw116 + (size_t)e * kEH * kD;
    const _Float16* w2 = exw216 + (size_t)e * kMO * kEH;
    const float* b1 = PP(EX_B1) + (size_t)e * kEH;
    const float* b2 = PP(EX_B2) + (size_t)e * kMO;
    gemm(stream, mctx16, kD, w1, kD, b1, nullptr, nullptr, hidf16 + (size_t)e * kEH, kE * kEH, kB, kEH, kD, 1);
    gemm(stream, hidf16 + (size_t)e * kEH, kE * kEH, w2, kEH, b2, nullptr, allo_f + (size_t)e * kMO, nullptr, kE * kMO, kB, kMO, kEH, 0);
    gemm(stream, ctxm16, kD, w1, kD, b1, nullptr, nullptr, hidr16 + (size_t)e * kEH, kE * kEH, kN, kEH, kD, 1);
    gemm(stream, hidr16 + (size_t)e * kEH, kE * kEH, w2, kEH, b2, nullptr, allo_r + (size_t)e * kMO, nullptr, kE * kMO, kN, kMO, kEH, 0);
  }

  // ---- 9. top-2 combines + aux stats ----
  k_moe_combine<<<kB, kMO, 0, stream>>>(lg_f, allo_f, y_f, nullptr);
  k_moe_combine<<<kB, kMO, 0, stream>>>(lg_l, allo_f, nullptr, yfl16);
  k_moe_combine<<<kN, kMO, 0, stream>>>(lg_r, allo_r, y_r, nullptr);
  k_gate_stats<<<1, 256, 0, stream>>>(lg_f, kB, aux6 + 0);
  k_gate_stats<<<1, 256, 0, stream>>>(lg_l, kB, aux6 + 2);
  k_gate_stats<<<1, 256, 0, stream>>>(lg_r, kN, aux6 + 4);

  // ---- 10. heads ----
  k_predin<<<(kN * 320 + 255) / 256, 256, 0, stream>>>(pooled, ctxm, y_f, predin16);
  gemm(stream, predin16, 320, phw16, 320, PP(PH_B), nullptr, out_pred, nullptr, 8, kN, 8, 320, 0);
  gemm(stream, yfl16, kMO, hzw16, kMO, PP(HZ_B), nullptr, out_haz, nullptr, 64, kB, 64, kMO, 2);
  k_rcain<<<(kN * 352 + 255) / 256, 256, 0, stream>>>(pooled, ctxm, y_r, x, lastv, rcain16);
  gemm(stream, rcain16, 352, rcw16, 352, PP(RC_B), nullptr, out_rca, nullptr, 1, kN, 1, 352, 0);
  k_finalize<<<1, 1, 0, stream>>>(aux6, out_aux);
}
